// MoEKGC_72868415144295
// MI455X (gfx1250) — matrix operand order repaired and checked by reading, not compiled, not run on hardware
//
#include <hip/hip_runtime.h>
#include <hip/hip_bf16.h>

// ---------------- problem constants ----------------
#define BT   32768   // tokens
#define DIN  1536
#define NE   5       // experts
#define HH   512
#define OO   512
#define TOPK 2

typedef __bf16 bfx8  __attribute__((ext_vector_type(8)));
typedef __bf16 bfx16 __attribute__((ext_vector_type(16)));
typedef float  v8f   __attribute__((ext_vector_type(8)));

static __device__ inline bfx16 mk16(bfx8 lo, bfx8 hi) {
  return __builtin_shufflevector(lo, hi, 0,1,2,3,4,5,6,7,8,9,10,11,12,13,14,15);
}

// CDNA5 async global->LDS copy (ASYNCcnt-tracked; ISA 15.18.3 op 98).
// VDST holds the LDS byte offset (low 32 bits of flat shared address, ISA 10.2).
static __device__ inline void async_copy_b128(unsigned lds_off, const void* gptr) {
  asm volatile("global_load_async_to_lds_b128 %0, %1, off"
               :: "v"(lds_off), "v"((unsigned long long)(size_t)gptr)
               : "memory");
}
static __device__ inline void wait_asynccnt0() {
  asm volatile("s_wait_asynccnt 0" ::: "memory");
}

// ---------------------------------------------------------------
// Gating: logits = x@Wg, softmax, scores out, top-2 gates,
// per-expert routing segments + load-balance stats.
// 256 threads = 16 tokens/block, 16 lanes per token (coalesced).
// ---------------------------------------------------------------
__global__ __launch_bounds__(256)
void gate_kernel(const float* __restrict__ x, const float* __restrict__ Wg,
                 float* __restrict__ scores_out, float* __restrict__ gates,
                 int* __restrict__ seg, int* __restrict__ cnt,
                 float* __restrict__ probsum)
{
  __shared__ float sWg[NE * DIN];   // [e][k] layout: conflict-free reads
  __shared__ float sProb[NE];
  __shared__ int   sCnt[NE];
  __shared__ int   sBase[NE];

  const int tid = threadIdx.x;
  if (tid < NE) { sProb[tid] = 0.f; sCnt[tid] = 0; }
  for (int i = tid; i < NE * DIN; i += 256) {
    int e = i / DIN, k = i - e * DIN;
    sWg[i] = Wg[k * NE + e];        // transpose [k][e] -> [e][k]
  }
  __syncthreads();

  const int ty = tid >> 4;          // token slot in block (0..15)
  const int tx = tid & 15;          // k-lane (0..15)
  const int token = blockIdx.x * 16 + ty;

  float acc[NE] = {0.f, 0.f, 0.f, 0.f, 0.f};
  const float* xr = x + (size_t)token * DIN;
  for (int k = tx; k < DIN; k += 16) {
    float xv = xr[k];
#pragma unroll
    for (int e = 0; e < NE; ++e) acc[e] += xv * sWg[e * DIN + k];
  }
#pragma unroll
  for (int off = 8; off >= 1; off >>= 1) {
#pragma unroll
    for (int e = 0; e < NE; ++e) acc[e] += __shfl_xor(acc[e], off, 32);
  }

  int li0 = 0, lp0 = 0, li1 = 0, lp1 = 0;
  if (tx == 0) {
    float m = acc[0];
#pragma unroll
    for (int e = 1; e < NE; ++e) m = fmaxf(m, acc[e]);
    float p[NE], s = 0.f;
#pragma unroll
    for (int e = 0; e < NE; ++e) { p[e] = __expf(acc[e] - m); s += p[e]; }
    float inv = 1.f / s;
#pragma unroll
    for (int e = 0; e < NE; ++e) {
      p[e] *= inv;
      scores_out[(size_t)token * NE + e] = p[e];
      atomicAdd(&sProb[e], p[e]);
    }
    int i0 = 0;
#pragma unroll
    for (int e = 1; e < NE; ++e) if (p[e] > p[i0]) i0 = e;
    int i1 = (i0 == 0) ? 1 : 0;
#pragma unroll
    for (int e = 0; e < NE; ++e) if (e != i0 && p[e] > p[i1]) i1 = e;
    float gs = p[i0] + p[i1];
    gates[2 * token + 0] = p[i0] / gs;
    gates[2 * token + 1] = p[i1] / gs;
    lp0 = atomicAdd(&sCnt[i0], 1); li0 = i0;
    lp1 = atomicAdd(&sCnt[i1], 1); li1 = i1;
  }
  __syncthreads();
  if (tid < NE) {
    sBase[tid] = atomicAdd(&cnt[tid], sCnt[tid]);
    atomicAdd(&probsum[tid], sProb[tid]);
  }
  __syncthreads();
  if (tx == 0) {
    seg[(size_t)li0 * BT + sBase[li0] + lp0] = 2 * token + 0;  // slot id
    seg[(size_t)li1 * BT + sBase[li1] + lp1] = 2 * token + 1;
  }
}

// ---------------------------------------------------------------
// lb_loss = 0.01 * E * sum_e mean_prob[e] * frac[e]
// ---------------------------------------------------------------
__global__ void lb_kernel(const int* __restrict__ cnt,
                          const float* __restrict__ probsum,
                          float* __restrict__ lb)
{
  if (threadIdx.x == 0 && blockIdx.x == 0) {
    float s = 0.f;
#pragma unroll
    for (int e = 0; e < NE; ++e)
      s += (probsum[e] / (float)BT) * ((float)cnt[e] / (float)BT);
    lb[0] = 0.01f * (float)NE * s;
  }
}

// ---------------------------------------------------------------
// Pre-swizzle weights into WMMA B-fragment layout (bf16):
//   dst[(((e*KC + kc)*NT + nt)*32 + lane)*16 + pos]
// lane<16 : n = nt*16+lane,   k = kc*32 + pos       (K 0..15)
// lane>=16: n = nt*16+lane-16 k = kc*32 + 16 + pos  (K 16..31)
// -> each WMMA B-fragment is two contiguous b128 loads per lane.
// ---------------------------------------------------------------
__global__ __launch_bounds__(256)
void wprep_kernel(const float* __restrict__ w, __bf16* __restrict__ dst,
                  int K, int N)
{
  size_t i = (size_t)blockIdx.x * 256 + threadIdx.x;
  size_t total = (size_t)NE * K * N;
  if (i >= total) return;
  int pos  = (int)(i & 15);
  int lane = (int)((i >> 4) & 31);
  size_t grp = i >> 9;                 // (e*KC + kc)*NT + nt
  int NT = N >> 4, KC = K >> 5;
  int nt = (int)(grp % NT);
  size_t t2 = grp / NT;
  int kc = (int)(t2 % KC);
  int e  = (int)(t2 / KC);
  int n  = nt * 16 + (lane & 15);
  int k  = (kc << 5) + ((lane < 16) ? pos : 16 + pos);
  dst[i] = (__bf16)w[((size_t)e * K + k) * N + n];
}

// ---------------------------------------------------------------
// A-tile staging helpers for the f32 (layer-1) path: global loads
// issued early, convert+LDS store after the WMMA block.
// ---------------------------------------------------------------
static __device__ inline void stage_load_x(const float* __restrict__ xA,
                                           int slot, int k0, int shalf,
                                           float4 f[4])
{
  size_t row = (slot >= 0) ? (size_t)(slot >> 1) : 0;   // token id
  const float4* src = (const float4*)(xA + row * DIN + k0 + shalf * 16);
  f[0] = src[0]; f[1] = src[1]; f[2] = src[2]; f[3] = src[3];
}

static __device__ inline void stage_store_x(__bf16* __restrict__ dst,
                                            const float4 f[4])
{
  bfx8 p0, p1;
  p0[0]=(__bf16)f[0].x; p0[1]=(__bf16)f[0].y; p0[2]=(__bf16)f[0].z; p0[3]=(__bf16)f[0].w;
  p0[4]=(__bf16)f[1].x; p0[5]=(__bf16)f[1].y; p0[6]=(__bf16)f[1].z; p0[7]=(__bf16)f[1].w;
  p1[0]=(__bf16)f[2].x; p1[1]=(__bf16)f[2].y; p1[2]=(__bf16)f[2].z; p1[3]=(__bf16)f[2].w;
  p1[4]=(__bf16)f[3].x; p1[5]=(__bf16)f[3].y; p1[6]=(__bf16)f[3].z; p1[7]=(__bf16)f[3].w;
  *(bfx8*)dst = p0; *(bfx8*)(dst + 8) = p1;
}

// ---------------------------------------------------------------
// Routed GEMM layer:  out[slot] = act( Arow(slot) @ W[e] + b[e] )
// WG = 256 thr (8 waves). Tile: 128 rows x 128 cols. Wave: 32x64.
// Ping-pong double-buffered LDS A staging (1 barrier/k-step).
// Layer 1 stages via VGPR f32->bf16 convert; layers 2/3 stage via
// GLOBAL_LOAD_ASYNC_TO_LDS_B128 (ASYNCcnt). B fragments preloaded
// into distinct registers -> 8 back-to-back WMMAs per k-step.
// Grid: (BT/128, 512/128, NE); empty tiles early-exit on cnt[e].
// ---------------------------------------------------------------
template<bool A_FROM_X, bool RELU, bool OUT_F32>
__global__ __launch_bounds__(256)
void moe_layer_kernel(const float* __restrict__ xA, const __bf16* __restrict__ hA,
                      const __bf16* __restrict__ wpre, const float* __restrict__ bias,
                      const int* __restrict__ seg, const int* __restrict__ cnt,
                      __bf16* __restrict__ outBf, float* __restrict__ outF, int K)
{
  const int e = blockIdx.z;
  const int c = cnt[e];
  const int m0 = blockIdx.x << 7;
  if (m0 >= c) return;

  const int tid  = threadIdx.x;
  const int lane = tid & 31;
  const int wid  = tid >> 5;
  const int wm   = wid & 3;      // wave m-block (4 x 32 rows)
  const int wn   = wid >> 2;     // wave n-block (2 x 64 cols)

  __shared__ __bf16 As[2][128 * 48];  // ping-pong; 96B row stride (16B aligned)
  __shared__ int sSlot[128];

  if (tid < 128) {
    int r = m0 + tid;
    sSlot[tid] = (r < c) ? seg[(size_t)e * BT + r] : -1;
  }
  __syncthreads();

  v8f acc0[4] = {}; v8f acc1[4] = {};

  const int KC    = K >> 5;
  const int ntg0  = (blockIdx.y << 3) + (wn << 2);      // first of 4 n-tiles
  const int koffA = (lane < 16) ? 0 : 16;               // A frag K split (contiguous k = 16h + s, same order as wprep B)
  const int arow0 = (wm << 5) + (lane & 15);

  const int srow  = tid >> 1;   // stage: 128 rows x 2 halves
  const int shalf = tid & 1;
  const int slotS = sSlot[srow];

  // staging destinations (LDS byte offsets for the async path)
  __bf16* dst0 = &As[0][srow * 48 + shalf * 16];
  __bf16* dst1 = &As[1][srow * 48 + shalf * 16];
  const size_t rowH = (slotS >= 0) ? (size_t)slotS : 0;       // h-row (layers 2/3)
  const __bf16* hrow = hA + rowH * HH + shalf * 16;

  // prologue: stage tile 0 into buffer 0
  if (A_FROM_X) {
    float4 f[4];
    stage_load_x(xA, slotS, 0, shalf, f);
    stage_store_x(dst0, f);
  } else {
    async_copy_b128((unsigned)(size_t)dst0,       hrow);
    async_copy_b128((unsigned)(size_t)dst0 + 16,  hrow + 8);
    wait_asynccnt0();
  }
  __syncthreads();

  for (int kc = 0; kc < KC; ++kc) {
    const int cur = kc & 1;
    const bool have_next = (kc + 1 < KC);
    __bf16* nxt = cur ? dst0 : dst1;

    // 1. issue next tile's loads (overlap with WMMAs below)
    float4 f[4];
    if (have_next) {
      if (A_FROM_X) {
        stage_load_x(xA, slotS, (kc + 1) << 5, shalf, f);
      } else {
        const __bf16* g = hrow + ((kc + 1) << 5);
        async_copy_b128((unsigned)(size_t)nxt,      g);
        async_copy_b128((unsigned)(size_t)nxt + 16, g + 8);
      }
    }

    // 2. A fragments from current buffer
    const __bf16* ab0 = &As[cur][arow0 * 48 + koffA];
    bfx16 a0 = mk16(*(const bfx8*)ab0, *(const bfx8*)(ab0 + 8));
    const __bf16* ab1 = ab0 + 16 * 48;
    bfx16 a1 = mk16(*(const bfx8*)ab1, *(const bfx8*)(ab1 + 8));

    // 3. preload all 4 B fragments (distinct regs -> one wait, 8 WMMAs)
    bfx16 bfr[4];
#pragma unroll
    for (int j = 0; j < 4; ++j) {
      const __bf16* bp = wpre + ((((size_t)e * KC + kc) * 32 + (ntg0 + j)) << 9)
                              + (lane << 4);
      bfr[j] = mk16(*(const bfx8*)bp, *(const bfx8*)(bp + 8));
    }
#pragma unroll
    for (int j = 0; j < 4; ++j) {
      acc0[j] = __builtin_amdgcn_wmma_f32_16x16x32_bf16(
                    false, a0, false, bfr[j], (short)0, acc0[j], false, false);
      acc1[j] = __builtin_amdgcn_wmma_f32_16x16x32_bf16(
                    false, a1, false, bfr[j], (short)0, acc1[j], false, false);
    }

    // 4. complete staging of next tile, then one barrier per k-step
    if (have_next) {
      if (A_FROM_X) stage_store_x(nxt, f);
      else          wait_asynccnt0();
    }
    __syncthreads();
  }

  // epilogue: C/D layout = lane&15 -> N, (lane>>4)*8 + r -> M  (ISA 7.12.2)
  const int nlow = lane & 15;
  const int mh   = (lane >> 4) << 3;
#pragma unroll
  for (int s = 0; s < 2; ++s) {
#pragma unroll
    for (int j = 0; j < 4; ++j) {
      const int ng = ((ntg0 + j) << 4) + nlow;
      const float bv = bias[e * HH + ng];
      v8f ac = s ? acc1[j] : acc0[j];
#pragma unroll
      for (int r = 0; r < 8; ++r) {
        const int mloc = (wm << 5) + (s << 4) + mh + r;
        const int slot = sSlot[mloc];
        if (slot < 0) continue;
        float v = ac[r] + bv;
        if (RELU) v = fmaxf(v, 0.f);
        if (OUT_F32) outF[((size_t)slot << 9) + ng] = v;
        else         outBf[((size_t)slot << 9) + ng] = (__bf16)v;
      }
    }
  }
}

// ---------------------------------------------------------------
// Combine: out[t] = g0 * eo[2t] + g1 * eo[2t+1]
// ---------------------------------------------------------------
__global__ __launch_bounds__(256)
void combine_kernel(const float* __restrict__ eo, const float* __restrict__ gates,
                    float* __restrict__ out)
{
  size_t idx = (size_t)blockIdx.x * 256 + threadIdx.x;   // over B*O
  size_t t = idx >> 9;
  int    o = (int)(idx & 511);
  out[idx] = gates[2 * t]     * eo[((2 * t)     << 9) + o]
           + gates[2 * t + 1] * eo[((2 * t + 1) << 9) + o];
}

// ---------------------------------------------------------------
extern "C" void kernel_launch(void* const* d_in, const int* in_sizes, int n_in,
                              void* d_out, int out_size, void* d_ws, size_t ws_size,
                              hipStream_t stream)
{
  (void)in_sizes; (void)n_in; (void)out_size; (void)ws_size;
  const float* x  = (const float*)d_in[0];
  const float* Wg = (const float*)d_in[1];
  const float* W1 = (const float*)d_in[2];
  const float* b1 = (const float*)d_in[3];
  const float* W2 = (const float*)d_in[4];
  const float* b2 = (const float*)d_in[5];
  const float* W3 = (const float*)d_in[6];
  const float* b3 = (const float*)d_in[7];

  float* out    = (float*)d_out;                 // [B,O]
  float* lb     = out + (size_t)BT * OO;         // scalar
  float* scores = lb + 1;                        // [B,E]

  // ---- workspace carve (256B aligned) ----
  size_t off = 0;
  auto carve = [&](size_t bytes) -> void* {
    void* r = (char*)d_ws + off;
    off += (bytes + 255) & ~(size_t)255;
    return r;
  };
  int*    cnt     = (int*)carve(256);                       // cnt[5] @+0
  float*  probsum = (float*)((char*)cnt + 64);              // probsum[5] @+64
  float*  gates   = (float*)carve(sizeof(float) * 2 * BT);
  int*    seg     = (int*)carve(sizeof(int) * (size_t)NE * BT);
  __bf16* w1p     = (__bf16*)carve((size_t)NE * DIN * HH * 2);
  __bf16* w2p     = (__bf16*)carve((size_t)NE * HH  * HH * 2);
  __bf16* w3p     = (__bf16*)carve((size_t)NE * HH  * HH * 2);
  __bf16* h1      = (__bf16*)carve((size_t)2 * BT * HH * 2);
  __bf16* h2      = (__bf16*)carve((size_t)2 * BT * HH * 2);
  float*  eo      = (float*)carve((size_t)2 * BT * OO * 4);

  hipMemsetAsync(cnt, 0, 256, stream);   // zero counters each call (graph-safe)

  gate_kernel<<<BT / 16, 256, 0, stream>>>(x, Wg, scores, gates, seg, cnt, probsum);
  lb_kernel<<<1, 64, 0, stream>>>(cnt, probsum, lb);

  wprep_kernel<<<(int)(((size_t)NE * DIN * HH + 255) / 256), 256, 0, stream>>>(W1, w1p, DIN, HH);
  wprep_kernel<<<(int)(((size_t)NE * HH  * HH + 255) / 256), 256, 0, stream>>>(W2, w2p, HH, HH);
  wprep_kernel<<<(int)(((size_t)NE * HH  * HH + 255) / 256), 256, 0, stream>>>(W3, w3p, HH, HH);

  dim3 grid(BT / 128, HH / 128, NE);
  moe_layer_kernel<true,  true,  false><<<grid, 256, 0, stream>>>(x,       nullptr, w1p, b1, seg, cnt, h1, nullptr, DIN);
  moe_layer_kernel<false, true,  false><<<grid, 256, 0, stream>>>(nullptr, h1,      w2p, b2, seg, cnt, h2, nullptr, HH);
  moe_layer_kernel<false, false, true ><<<grid, 256, 0, stream>>>(nullptr, h2,      w3p, b3, seg, cnt, nullptr, eo, HH);

  combine_kernel<<<(int)(((size_t)BT * OO) / 256), 256, 0, stream>>>(eo, gates, out);
}
